// BlockXDiag_90142773608651
// MI455X (gfx1250) — matrix operand order repaired and checked by reading, not compiled, not run on hardware
//
#include <hip/hip_runtime.h>
#include <stdint.h>

// ---------------------------------------------------------------------------
// BlockXDiag (tri-diagonal block matrix, periodic) for MI455X / gfx1250.
//  out[:, i] = x_{i-1} @ Wm(i).T + x_i @ Wd[i].T + x_{i+1} @ Wp(i).T
// Split-f16 WMMA (hi+lo planes) => fp32-grade accuracy on the matrix pipe.
// Both A and B operands are pre-split + pre-swizzled into WMMA fragment
// order in workspace, so the hot loop is pure ds_load_b128 + v_wmma.
// ---------------------------------------------------------------------------

#define MB   64      // blocks per row/col
#define PB   256     // block size
#define BATCH 4096
#define NMAT 192     // 64 Wd + 63 Wu + 63 Wl + Wtr + Wbl
#define MAT_ELEMS (PB * PB)          // 65536
#define ROWLEN (MB * PB)             // 16384

typedef __attribute__((ext_vector_type(16))) _Float16 v16h;
typedef __attribute__((ext_vector_type(8)))  float    v8f;
typedef __attribute__((__vector_size__(16))) int      v4i;
typedef __attribute__((address_space(1)))    v4i      gv4i;   // global int4
typedef __attribute__((address_space(3)))    v4i      lv4i;   // LDS int4

// ---------------- async global->LDS copy (guarded) -------------------------
#if defined(__has_builtin)
#  if __has_builtin(__builtin_amdgcn_global_load_async_to_lds_b128)
#    define HAVE_ASYNC 1
#  endif
#endif
#ifndef HAVE_ASYNC
#  define HAVE_ASYNC 0
#endif

__device__ __forceinline__ void cp16(void* dst_lds, const void* src_glb) {
#if HAVE_ASYNC
  __builtin_amdgcn_global_load_async_to_lds_b128(
      (gv4i*)src_glb, (lv4i*)dst_lds, 0, 0);
#else
  *(uint4*)dst_lds = *(const uint4*)src_glb;
#endif
}

__device__ __forceinline__ void wait_async() {
#if HAVE_ASYNC
#  if defined(__has_builtin) && __has_builtin(__builtin_amdgcn_s_wait_asynccnt)
  __builtin_amdgcn_s_wait_asynccnt(0);
#  else
  asm volatile("s_wait_asynccnt 0" ::: "memory");
#  endif
#endif
}

// A-fragment k mapping (contiguous, matches B): slot j -> k = 16*half + j within 32-chunk
__device__ __forceinline__ int kidx(int half, int j) {
  return 16 * half + j;
}

// ---------------------------------------------------------------------------
// Prep 1: weights -> transposed, B-fragment-ordered f16 hi/lo planes.
// Per matrix: group g = ((c*16 + nt)*2 + kh)*16 + nl, 16 contiguous f16:
//   value = W[n][k], n = nt*16+nl, k = c*32+kh*16+j.
// ---------------------------------------------------------------------------
__global__ void __launch_bounds__(256)
prep_weights(const float* __restrict__ Wd, const float* __restrict__ Wu,
             const float* __restrict__ Wl, const float* __restrict__ Wtr,
             const float* __restrict__ Wbl,
             _Float16* __restrict__ hi, _Float16* __restrict__ lo) {
  const int m = blockIdx.x;                       // matrix index 0..191
  const int g = blockIdx.y * 256 + threadIdx.x;   // group index 0..4095

  const float* src;
  if (m < 64)       src = Wd  + (size_t)m * MAT_ELEMS;
  else if (m < 127) src = Wu  + (size_t)(m - 64)  * MAT_ELEMS;
  else if (m < 190) src = Wl  + (size_t)(m - 127) * MAT_ELEMS;
  else if (m == 190) src = Wtr;
  else               src = Wbl;

  const int nl = g & 15;
  const int kh = (g >> 4) & 1;
  const int nt = (g >> 5) & 15;
  const int c  = g >> 9;                          // 0..7
  const int n  = nt * 16 + nl;
  const int k0 = c * 32 + kh * 16;
  const size_t dstBase = (size_t)m * MAT_ELEMS + (size_t)g * 16;

#pragma unroll
  for (int j = 0; j < 16; ++j) {
    float w = src[n * PB + k0 + j];
    _Float16 h = (_Float16)w;
    _Float16 l = (_Float16)(w - (float)h);
    hi[dstBase + j] = h;
    lo[dstBase + j] = l;
  }
}

// ---------------------------------------------------------------------------
// Prep 2: x -> A-fragment-ordered f16 hi/lo planes.
// Plane layout: index = ((mt*512 + kc)*32 + L)*16 + j
//   value = x[mt*16 + (L&15)][kc*32 + kidx(L>>4, j)]
// (mt = 16-row tile 0..255, kc = 32-col chunk 0..511)
// Coalesced via an LDS-staged 16x512 fp32 tile per block.
// ---------------------------------------------------------------------------
#define TSTR 516
__global__ void __launch_bounds__(256)
prep_x(const float* __restrict__ x,
       _Float16* __restrict__ xhi, _Float16* __restrict__ xlo) {
  __shared__ alignas(16) float tile[16 * TSTR];   // ~33 KB
  const int mt  = blockIdx.x;                     // 0..255
  const int kcb = blockIdx.y;                     // 0..31  (16 kc each)
  const int tid = threadIdx.x;

  // load 16 rows x 512 cols fp32, fully coalesced
#pragma unroll
  for (int it = 0; it < 8; ++it) {
    const int f4  = it * 256 + tid;               // 0..2047
    const int r   = f4 >> 7;                      // 0..15
    const int c4  = f4 & 127;                     // 0..127
    const float4 v = *(const float4*)(x + (size_t)(mt * 16 + r) * ROWLEN
                                        + kcb * 512 + c4 * 4);
    *(float4*)&tile[r * TSTR + c4 * 4] = v;
  }
  __syncthreads();

  // emit fragment-ordered hi/lo groups (2 groups of 16 f16 per thread)
#pragma unroll
  for (int gg = 0; gg < 2; ++gg) {
    const int g    = tid * 2 + gg;                // 0..511
    const int L    = g & 31;
    const int kcl  = g >> 5;                      // 0..15
    const int half = L >> 4;
    const int r    = L & 15;
    const size_t dst = ((size_t)(mt * 512 + kcb * 16 + kcl) * 32 + L) * 16;
    v16h h, l;
#pragma unroll
    for (int j = 0; j < 16; ++j) {
      float w = tile[r * TSTR + kcl * 32 + kidx(half, j)];
      _Float16 hh = (_Float16)w;
      h[j] = hh;
      l[j] = (_Float16)(w - (float)hh);
    }
    *(v16h*)(xhi + dst) = h;
    *(v16h*)(xlo + dst) = l;
  }
}

// ---------------------------------------------------------------------------
// Shared index helpers for the main kernels
// ---------------------------------------------------------------------------
__device__ __forceinline__ int srcBlkOf(int i, int role) {
  return (role == 0) ? ((i + MB - 1) & (MB - 1))
       : (role == 1) ? i : ((i + 1) & (MB - 1));
}
__device__ __forceinline__ int matOf(int i, int role) {
  return (role == 1) ? i
       : (role == 0) ? ((i == 0)      ? 190 : 126 + i)
                     : ((i == MB - 1) ? 191 : 64 + i);
}

// ---------------------------------------------------------------------------
// Main kernel, pre-split path: A and B both f16 fragment planes in ws.
// Grid (64 batch-tiles, 64 i, 2 n-halves); 256 threads = 8 waves.
// Hot loop: ds_load_b128 fragments + 12 v_wmma per chunk, no VALU convert.
// ---------------------------------------------------------------------------
__global__ void __launch_bounds__(256)
bxdiag_main_pre(const _Float16* __restrict__ xHi, const _Float16* __restrict__ xLo,
                const _Float16* __restrict__ wHi, const _Float16* __restrict__ wLo,
                float* __restrict__ out) {
  __shared__ alignas(32) _Float16 lA[2][2][2048];  //  16 KB (buf, hi/lo)
  __shared__ alignas(32) _Float16 lB[2][2][4096];  //  32 KB

  const int bt   = blockIdx.x;
  const int i    = blockIdx.y;
  const int nh   = blockIdx.z;
  const int tid  = threadIdx.x;
  const int lane = tid & 31;
  const int wave = tid >> 5;
  const int tileM = wave & 3;
  const int waveN = wave >> 2;
  const int nl   = lane & 15;
  const int half = lane >> 4;

  v8f acc[4] = {};

  auto stage = [&](int c, int buf) {
    const int role = c >> 3;
    const int cm   = c & 7;
    const int kc   = srcBlkOf(i, role) * 8 + cm;   // global k-chunk in x row
    const int mat  = matOf(i, role);
    // A: 4 row-tiles x 512 f16 per plane
    {
      const int t4  = tid >> 6;                    // 0..3
      const int off = (tid & 63) * 8;              // 0..504
      const size_t sb = ((size_t)((bt * 4 + t4) * 512 + kc) * 512) + off;
      cp16(&lA[buf][0][t4 * 512 + off], xHi + sb);
      cp16(&lA[buf][1][t4 * 512 + off], xLo + sb);
    }
    // B: 4096 f16 per plane (fragment-ordered, contiguous)
    {
      const int off = tid * 16;
      const size_t sb = (size_t)mat * MAT_ELEMS + cm * 8192 + nh * 4096 + off;
      cp16(&lB[buf][0][off],     wHi + sb);
      cp16(&lB[buf][0][off + 8], wHi + sb + 8);
      cp16(&lB[buf][1][off],     wLo + sb);
      cp16(&lB[buf][1][off + 8], wLo + sb + 8);
    }
  };

  stage(0, 0);
  int buf = 0;

  for (int c = 0; c < 24; ++c) {
    wait_async();
    __syncthreads();
    if (c + 1 < 24) stage(c + 1, buf ^ 1);

    const int aoff = tileM * 512 + lane * 16;
    v16h ah = *(const v16h*)&lA[buf][0][aoff];
    v16h al = *(const v16h*)&lA[buf][1][aoff];

#pragma unroll
    for (int nt = 0; nt < 4; ++nt) {
      const int eo = ((((waveN * 4 + nt) * 2 + half) * 16) + nl) * 16;
      v16h bh = *(const v16h*)&lB[buf][0][eo];
      v16h bl = *(const v16h*)&lB[buf][1][eo];
      acc[nt] = __builtin_amdgcn_wmma_f32_16x16x32_f16(
          false, ah, false, bh, (short)0, acc[nt], false, false);
      acc[nt] = __builtin_amdgcn_wmma_f32_16x16x32_f16(
          false, ah, false, bl, (short)0, acc[nt], false, false);
      acc[nt] = __builtin_amdgcn_wmma_f32_16x16x32_f16(
          false, al, false, bh, (short)0, acc[nt], false, false);
    }
    __syncthreads();
    buf ^= 1;
  }

#pragma unroll
  for (int nt = 0; nt < 4; ++nt) {
    const int col = i * PB + nh * 128 + waveN * 64 + nt * 16 + nl;
#pragma unroll
    for (int r = 0; r < 8; ++r) {
      const int row = bt * 64 + tileM * 16 + 8 * half + r;
      out[(size_t)row * ROWLEN + col] = acc[nt][r];
    }
  }
}

// ---------------------------------------------------------------------------
// Fallback path (small workspace): A staged as fp32, split in-register.
// ---------------------------------------------------------------------------
__global__ void __launch_bounds__(256)
bxdiag_main_cvt(const float* __restrict__ x,
                const _Float16* __restrict__ wHi, const _Float16* __restrict__ wLo,
                float* __restrict__ out) {
  __shared__ alignas(16) float    lA[2][64 * 36];
  __shared__ alignas(32) _Float16 lB[2][2][4096];

  const int bt   = blockIdx.x;
  const int i    = blockIdx.y;
  const int nh   = blockIdx.z;
  const int tid  = threadIdx.x;
  const int lane = tid & 31;
  const int wave = tid >> 5;
  const int tileM = wave & 3;
  const int waveN = wave >> 2;
  const int nl   = lane & 15;
  const int half = lane >> 4;

  v8f acc[4] = {};

  auto stage = [&](int c, int buf) {
    const int role = c >> 3;
    const int cm   = c & 7;
    const int srcBlk = srcBlkOf(i, role);
    const int mat    = matOf(i, role);
    {
      const int r = tid >> 2;
      const int q = (tid & 3) * 8;
      const float* src = x + (size_t)(bt * 64 + r) * ROWLEN
                           + srcBlk * PB + cm * 32 + q;
      float* dst = &lA[buf][r * 36 + q];
      cp16(dst, src);
      cp16(dst + 4, src + 4);
    }
    {
      const int off = tid * 16;
      const size_t sb = (size_t)mat * MAT_ELEMS + cm * 8192 + nh * 4096 + off;
      cp16(&lB[buf][0][off],     wHi + sb);
      cp16(&lB[buf][0][off + 8], wHi + sb + 8);
      cp16(&lB[buf][1][off],     wLo + sb);
      cp16(&lB[buf][1][off + 8], wLo + sb + 8);
    }
  };

  stage(0, 0);
  int buf = 0;

  for (int c = 0; c < 24; ++c) {
    wait_async();
    __syncthreads();
    if (c + 1 < 24) stage(c + 1, buf ^ 1);

    const float* ar = &lA[buf][(tileM * 16 + nl) * 36];
    v16h ah, al;
#pragma unroll
    for (int j = 0; j < 16; ++j) {
      float w = ar[kidx(half, j)];
      _Float16 h = (_Float16)w;
      ah[j] = h;
      al[j] = (_Float16)(w - (float)h);
    }

#pragma unroll
    for (int nt = 0; nt < 4; ++nt) {
      const int eo = ((((waveN * 4 + nt) * 2 + half) * 16) + nl) * 16;
      v16h bh = *(const v16h*)&lB[buf][0][eo];
      v16h bl = *(const v16h*)&lB[buf][1][eo];
      acc[nt] = __builtin_amdgcn_wmma_f32_16x16x32_f16(
          false, ah, false, bh, (short)0, acc[nt], false, false);
      acc[nt] = __builtin_amdgcn_wmma_f32_16x16x32_f16(
          false, ah, false, bl, (short)0, acc[nt], false, false);
      acc[nt] = __builtin_amdgcn_wmma_f32_16x16x32_f16(
          false, al, false, bh, (short)0, acc[nt], false, false);
    }
    __syncthreads();
    buf ^= 1;
  }

#pragma unroll
  for (int nt = 0; nt < 4; ++nt) {
    const int col = i * PB + nh * 128 + waveN * 64 + nt * 16 + nl;
#pragma unroll
    for (int r = 0; r < 8; ++r) {
      const int row = bt * 64 + tileM * 16 + 8 * half + r;
      out[(size_t)row * ROWLEN + col] = acc[nt][r];
    }
  }
}

// ---------------------------------------------------------------------------
extern "C" void kernel_launch(void* const* d_in, const int* in_sizes, int n_in,
                              void* d_out, int out_size, void* d_ws, size_t ws_size,
                              hipStream_t stream) {
  (void)in_sizes; (void)n_in; (void)out_size;
  const float* x   = (const float*)d_in[0];
  const float* Wd  = (const float*)d_in[1];
  const float* Wu  = (const float*)d_in[2];
  const float* Wl  = (const float*)d_in[3];
  const float* Wtr = (const float*)d_in[4];
  const float* Wbl = (const float*)d_in[5];
  float* out = (float*)d_out;

  const size_t wPlane = (size_t)NMAT * MAT_ELEMS;        // f16 elems per plane
  const size_t xPlane = (size_t)BATCH * ROWLEN;          // f16 elems per plane
  const size_t needW   = wPlane * 2 * sizeof(_Float16);  //  48 MB
  const size_t needPre = needW + xPlane * 2 * sizeof(_Float16); // +256 MB

  _Float16* wHi = (_Float16*)d_ws;
  _Float16* wLo = wHi + wPlane;

  prep_weights<<<dim3(NMAT, 16), 256, 0, stream>>>(Wd, Wu, Wl, Wtr, Wbl, wHi, wLo);

  if (ws_size >= needPre) {
    _Float16* xHi = (_Float16*)((char*)d_ws + needW);
    _Float16* xLo = xHi + xPlane;
    prep_x<<<dim3(BATCH / 16, 32), 256, 0, stream>>>(x, xHi, xLo);
    bxdiag_main_pre<<<dim3(BATCH / 64, MB, 2), 256, 0, stream>>>(xHi, xLo, wHi, wLo, out);
  } else {
    bxdiag_main_cvt<<<dim3(BATCH / 64, MB, 2), 256, 0, stream>>>(x, wHi, wLo, out);
  }
}
